// SimpleAttention_52750788330022
// MI455X (gfx1250) — compile-verified, matrix operand order checked by reading, not run on hardware
//
#include <hip/hip_runtime.h>
#include <hip/hip_bf16.h>

typedef __attribute__((ext_vector_type(4)))  __bf16 v4bf;
typedef __attribute__((ext_vector_type(8)))  __bf16 v8bf;
typedef __attribute__((ext_vector_type(16))) __bf16 v16bf;
typedef __attribute__((ext_vector_type(8)))  float  v8f;

#define EMBED  1024
#define S_LEN  1024
#define NBATCH 8
#define HEADS  16
#define HDIM   64

// Wait for all outstanding async (LDS<->memory) transfers.
#define ASYNC_WAIT0() asm volatile("s_wait_asynccnt 0x0" ::: "memory")

// Load one 16x32 bf16 A/B WMMA fragment for this lane.
// CDNA5 16-bit A/B layout: a lane holds 16 bf16 values as two contiguous
// 8-element (16-byte) groups separated by 16 elements:
//   lanes 0-15 : k = {0..7, 16..23}, lanes 16-31 : k = {8..15, 24..31}
// Caller bakes the half*8 offset into `p`. Works for global or LDS pointers.
__device__ __forceinline__ v16bf load_frag(const __bf16* p) {
    v8bf lo = *(const v8bf*)(p);
    v8bf hi = *(const v8bf*)(p + 16);
    return __builtin_shufflevector(lo, hi, 0,1,2,3,4,5,6,7,8,9,10,11,12,13,14,15);
}

__device__ __forceinline__ v8f wmma_bf16(v16bf a, v16bf b, v8f c) {
    // (neg_a, A, neg_b, B, c_mod, C, reuse_a, reuse_b)
    return __builtin_amdgcn_wmma_f32_16x16x32_bf16(false, a, false, b, (short)0, c,
                                                   false, false);
}

// ---------------------------------------------------------------------------
// f32 -> bf16 conversion, 4 elements / thread
// ---------------------------------------------------------------------------
__global__ __launch_bounds__(256) void cvt_f32_bf16(const float* __restrict__ in,
                                                    __bf16* __restrict__ out, int n4) {
    int i = blockIdx.x * blockDim.x + threadIdx.x;
    if (i < n4) {
        float4 f = ((const float4*)in)[i];
        v4bf o = { (__bf16)f.x, (__bf16)f.y, (__bf16)f.z, (__bf16)f.w };
        ((v4bf*)out)[i] = o;
    }
}

// ---------------------------------------------------------------------------
// C[M,N] = A[M,K] @ B[N,K]^T   (torch Linear semantics, K contiguous in both)
// 256 threads = 8 waves; block tile 128x128; wave tile 32x64 = 2x4 WMMA frags.
// A/B 128x32 k-tiles are staged in LDS with global_load_async_to_lds_b128,
// double-buffered (issue k+1, compute k, s_wait_asynccnt + barrier).
// LDS rows padded to 40 bf16 (80B) for conflict-free ds_load_b128.
// MODE 0: store bf16 row-major            (Q, K)
// MODE 1: store bf16 transposed-per-batch [n, col, s]  (V -> Vt)
// MODE 2: store f32 row-major + bias      (final projection)
// ---------------------------------------------------------------------------
#define LDS_PITCH 40

__device__ __forceinline__ void stage_tile_async(const __bf16* __restrict__ gbase,
                                                 int K, int k0,
                                                 __bf16* lds, int tid) {
    // 128 rows x 32 cols = 512 chunks of 16B; each thread issues 2 async copies.
#pragma unroll
    for (int u = 0; u < 2; u++) {
        const int id  = tid + u * 256;
        const int row = id >> 2;
        const int c   = id & 3;
        const unsigned ldsAddr =
            (unsigned)(uintptr_t)(lds + row * LDS_PITCH + c * 8);
        const unsigned long long ga =
            (unsigned long long)(uintptr_t)(gbase + (size_t)row * K + k0 + c * 8);
        asm volatile("global_load_async_to_lds_b128 %0, %1, off"
                     :: "v"(ldsAddr), "v"(ga) : "memory");
    }
}

template <int MODE>
__global__ __launch_bounds__(256) void gemm_bf16_k(
    const __bf16* __restrict__ A, const __bf16* __restrict__ B,
    void* __restrict__ C, const float* __restrict__ bias,
    int M, int N, int K) {
    __shared__ __bf16 ldsA[2][128 * LDS_PITCH];
    __shared__ __bf16 ldsB[2][128 * LDS_PITCH];

    const int tid   = threadIdx.x;
    const int lane  = tid & 31;
    const int wave  = tid >> 5;
    const int lm    = lane & 15;
    const int half  = lane >> 4;
    const int wm    = wave >> 1;   // 0..3
    const int wn    = wave & 1;    // 0..1

    const __bf16* Ablk = A + (size_t)(blockIdx.y * 128) * K;
    const __bf16* Bblk = B + (size_t)(blockIdx.x * 128) * K;

    v8f acc[2][4];
#pragma unroll
    for (int f = 0; f < 2; f++)
#pragma unroll
        for (int g = 0; g < 4; g++) acc[f][g] = (v8f){};

    // Per-lane fragment offsets inside the LDS tiles.
    int aOff[2], bOff[4];
#pragma unroll
    for (int f = 0; f < 2; f++)
        aOff[f] = (wm * 32 + f * 16 + lm) * LDS_PITCH + half * 8;
#pragma unroll
    for (int g = 0; g < 4; g++)
        bOff[g] = (wn * 64 + g * 16 + lm) * LDS_PITCH + half * 8;

    // Prologue: stage k-tile 0 into buffer 0.
    stage_tile_async(Ablk, K, 0, ldsA[0], tid);
    stage_tile_async(Bblk, K, 0, ldsB[0], tid);
    ASYNC_WAIT0();
    __syncthreads();

    const int nk = K / 32;
    for (int kt = 0; kt < nk; kt++) {
        const int cur = kt & 1;
        if (kt + 1 < nk) {   // issue next tile while computing this one
            stage_tile_async(Ablk, K, (kt + 1) * 32, ldsA[cur ^ 1], tid);
            stage_tile_async(Bblk, K, (kt + 1) * 32, ldsB[cur ^ 1], tid);
        }

        v16bf af[2], bfr[4];
#pragma unroll
        for (int f = 0; f < 2; f++) af[f] = load_frag(&ldsA[cur][aOff[f]]);
#pragma unroll
        for (int g = 0; g < 4; g++) bfr[g] = load_frag(&ldsB[cur][bOff[g]]);
#pragma unroll
        for (int f = 0; f < 2; f++)
#pragma unroll
            for (int g = 0; g < 4; g++)
                acc[f][g] = wmma_bf16(af[f], bfr[g], acc[f][g]);

        if (kt + 1 < nk) ASYNC_WAIT0();
        __syncthreads();
    }

    // C layout: VGPR r holds row (half ? r+8 : r) of the 16x16 tile, col = lm.
    const int mBase = blockIdx.y * 128 + wm * 32;
    const int nBase = blockIdx.x * 128 + wn * 64;
#pragma unroll
    for (int f = 0; f < 2; f++) {
#pragma unroll
        for (int g = 0; g < 4; g++) {
            const int col = nBase + g * 16 + lm;
#pragma unroll
            for (int r = 0; r < 8; r++) {
                const int row = mBase + f * 16 + half * 8 + r;
                const float v = acc[f][g][r];
                if (MODE == 0) {
                    ((__bf16*)C)[(size_t)row * N + col] = (__bf16)v;
                } else if (MODE == 1) {
                    const int b = row >> 10;          // batch (S = 1024)
                    const int s = row & 1023;
                    ((__bf16*)C)[((size_t)b * N + col) * S_LEN + s] = (__bf16)v;
                } else {
                    ((float*)C)[(size_t)row * N + col] = v + bias[col];
                }
            }
        }
    }
}

// ---------------------------------------------------------------------------
// Fused attention: one wave per (batch, head, 16 query rows), Sk chunk = 64.
//   scores = Q @ K^T * 1/sqrt(E); online softmax; O = P @ V
// Q, K in [n, s, h*64+d] bf16 (contiguous d), V transposed [n, h*64+d, s].
// Row-sums are computed as P @ ones via WMMA (result lands replicated in the
// same C-layout row mapping as the running max/sum state -> no shuffles).
// ---------------------------------------------------------------------------
#define P_PITCH 72   // 16x64 P tile padded to 72 (144B rows) -> conflict-free

__global__ __launch_bounds__(32) void attn_k(
    const __bf16* __restrict__ Q, const __bf16* __restrict__ Kmat,
    const __bf16* __restrict__ Vt, __bf16* __restrict__ ctx) {
    __shared__ __bf16 Plds[16][P_PITCH];

    const int lane = threadIdx.x & 31;
    const int lm   = lane & 15;
    const int half = lane >> 4;
    const int sq0  = blockIdx.x * 16;
    const int h    = blockIdx.y;
    const int b    = blockIdx.z;
    const float scale = 0.03125f;  // 1/sqrt(1024)

    v16bf ones;
#pragma unroll
    for (int i = 0; i < 16; i++) ones[i] = (__bf16)1.0f;

    // Q fragments: 16 rows x 64 (d) = two 16x32 A-frags, resident all loop.
    const size_t qRow = (size_t)(b * S_LEN + sq0 + lm) * EMBED + h * HDIM;
    v16bf qf[2];
    qf[0] = load_frag(Q + qRow + half * 8);
    qf[1] = load_frag(Q + qRow + 32 + half * 8);

    v8f o[4];
#pragma unroll
    for (int dc = 0; dc < 4; dc++) o[dc] = (v8f){};
    float mrow[8], lrow[8];
#pragma unroll
    for (int r = 0; r < 8; r++) { mrow[r] = -3.0e38f; lrow[r] = 0.0f; }

    for (int sk0 = 0; sk0 < S_LEN; sk0 += 64) {
        // ---- scores: four 16x16 C-frags covering 64 key columns ----
        v8f s[4];
#pragma unroll
        for (int nc = 0; nc < 4; nc++) {
            const size_t kRow =
                (size_t)(b * S_LEN + sk0 + nc * 16 + lm) * EMBED + h * HDIM;
            v16bf k0f = load_frag(Kmat + kRow + half * 8);
            v16bf k1f = load_frag(Kmat + kRow + 32 + half * 8);
            v8f c = (v8f){};
            c = wmma_bf16(qf[0], k0f, c);
            c = wmma_bf16(qf[1], k1f, c);
            s[nc] = c;
        }

        // ---- running max (16-lane half-wave shuffle reduce) + exp ----
        float alpha[8];
#pragma unroll
        for (int r = 0; r < 8; r++) {
            float a0 = s[0][r] * scale, a1 = s[1][r] * scale;
            float a2 = s[2][r] * scale, a3 = s[3][r] * scale;
            float cm = fmaxf(fmaxf(a0, a1), fmaxf(a2, a3));
#pragma unroll
            for (int m = 1; m < 16; m <<= 1) cm = fmaxf(cm, __shfl_xor(cm, m, 32));
            const float mn = fmaxf(mrow[r], cm);
            alpha[r] = __expf(mrow[r] - mn);
            mrow[r]  = mn;
            s[0][r] = __expf(a0 - mn);
            s[1][r] = __expf(a1 - mn);
            s[2][r] = __expf(a2 - mn);
            s[3][r] = __expf(a3 - mn);
        }
#pragma unroll
        for (int dc = 0; dc < 4; dc++)
#pragma unroll
            for (int r = 0; r < 8; r++) o[dc][r] *= alpha[r];

        // ---- re-layout P: C-layout -> A-fragments via LDS bounce ----
        __syncthreads();
#pragma unroll
        for (int r = 0; r < 8; r++) {
            const int row = half ? r + 8 : r;
#pragma unroll
            for (int nc = 0; nc < 4; nc++)
                Plds[row][nc * 16 + lm] = (__bf16)s[nc][r];
        }
        __syncthreads();
        const v16bf pf0 = load_frag(&Plds[lm][half * 8]);
        const v16bf pf1 = load_frag(&Plds[lm][32 + half * 8]);

        // ---- row sums via WMMA against all-ones B (no shuffles) ----
        v8f rs = (v8f){};
        rs = wmma_bf16(pf0, ones, rs);
        rs = wmma_bf16(pf1, ones, rs);
#pragma unroll
        for (int r = 0; r < 8; r++) lrow[r] = lrow[r] * alpha[r] + rs[r];

        // ---- O += P @ V : four 16x16 output d-chunks, two k-steps ----
#pragma unroll
        for (int dc = 0; dc < 4; dc++) {
            const size_t vRow =
                ((size_t)b * EMBED + h * HDIM + dc * 16 + lm) * S_LEN + sk0 + half * 8;
            o[dc] = wmma_bf16(pf0, load_frag(Vt + vRow), o[dc]);
            o[dc] = wmma_bf16(pf1, load_frag(Vt + vRow + 32), o[dc]);
        }
    }

    // ---- normalize & store context (bf16, natural [n, s, e]) ----
    float rl[8];
#pragma unroll
    for (int r = 0; r < 8; r++) rl[r] = 1.0f / lrow[r];
#pragma unroll
    for (int dc = 0; dc < 4; dc++) {
        const int col = h * HDIM + dc * 16 + lm;
#pragma unroll
        for (int r = 0; r < 8; r++) {
            const int row = sq0 + (half ? r + 8 : r);
            ctx[(size_t)(b * S_LEN + row) * EMBED + col] = (__bf16)(o[dc][r] * rl[r]);
        }
    }
}

// ---------------------------------------------------------------------------
extern "C" void kernel_launch(void* const* d_in, const int* in_sizes, int n_in,
                              void* d_out, int out_size, void* d_ws, size_t ws_size,
                              hipStream_t stream) {
    const float* x  = (const float*)d_in[0];
    const float* Wq = (const float*)d_in[1];
    const float* Wk = (const float*)d_in[2];
    const float* Wv = (const float*)d_in[3];
    const float* Wo = (const float*)d_in[4];
    const float* bo = (const float*)d_in[5];

    const size_t MB = (size_t)1 << 20;
    char* ws = (char*)d_ws;
    __bf16* xb  = (__bf16*)(ws);              // 16 MB
    __bf16* wqb = (__bf16*)(ws + 16 * MB);    //  2 MB
    __bf16* wkb = (__bf16*)(ws + 18 * MB);
    __bf16* wvb = (__bf16*)(ws + 20 * MB);
    __bf16* wob = (__bf16*)(ws + 22 * MB);
    __bf16* Qb  = (__bf16*)(ws + 24 * MB);    // 16 MB
    __bf16* Kb  = (__bf16*)(ws + 40 * MB);    // 16 MB
    __bf16* Vtb = (__bf16*)(ws + 56 * MB);    // 16 MB
    __bf16* Cb  = (__bf16*)(ws + 72 * MB);    // 16 MB  (total 88 MB)

    const int nx = NBATCH * S_LEN * EMBED;    // 8M
    const int nw = EMBED * EMBED;             // 1M
    cvt_f32_bf16<<<(nx / 4) / 256, 256, 0, stream>>>(x,  xb,  nx / 4);
    cvt_f32_bf16<<<(nw / 4) / 256, 256, 0, stream>>>(Wq, wqb, nw / 4);
    cvt_f32_bf16<<<(nw / 4) / 256, 256, 0, stream>>>(Wk, wkb, nw / 4);
    cvt_f32_bf16<<<(nw / 4) / 256, 256, 0, stream>>>(Wv, wvb, nw / 4);
    cvt_f32_bf16<<<(nw / 4) / 256, 256, 0, stream>>>(Wo, wob, nw / 4);

    const int Mrows = NBATCH * S_LEN;         // 8192
    dim3 gg(EMBED / 128, Mrows / 128);        // (8, 64)
    gemm_bf16_k<0><<<gg, 256, 0, stream>>>(xb, wqb, Qb,  nullptr, Mrows, EMBED, EMBED);
    gemm_bf16_k<0><<<gg, 256, 0, stream>>>(xb, wkb, Kb,  nullptr, Mrows, EMBED, EMBED);
    gemm_bf16_k<1><<<gg, 256, 0, stream>>>(xb, wvb, Vtb, nullptr, Mrows, EMBED, EMBED);

    attn_k<<<dim3(S_LEN / 16, HEADS, NBATCH), 32, 0, stream>>>(Qb, Kb, Vtb, Cb);

    gemm_bf16_k<2><<<gg, 256, 0, stream>>>(Cb, wob, d_out, bo, Mrows, EMBED, EMBED);
}
